// Attention_2894807958265
// MI455X (gfx1250) — matrix operand order repaired and checked by reading, not compiled, not run on hardware
//
#include <hip/hip_runtime.h>
#include <hip/hip_bf16.h>

typedef _Float16 v16h __attribute__((ext_vector_type(16)));
typedef _Float16 h8   __attribute__((ext_vector_type(8)));
typedef float    v8f  __attribute__((ext_vector_type(8)));

#define DIMX   512
#define HEADS  4
#define DH     128
#define SEQ    4096
#define BATCH  2
#define NOUT   1536            // 3 * HEADS * DH
#define ROWS   (BATCH * SEQ)   // 8192
#define SCALE_Q 0.08838834764831845f   // 128^-0.5

#define CH      32             // keys per attention chunk
#define KSTRIDE 136            // 128 + 8 halves pad (A-frag reads conflict-free)
#define VSTRIDE 40             // 32 + 8 halves pad

#if __has_builtin(__builtin_amdgcn_sched_barrier)
#define SCHED_FENCE() __builtin_amdgcn_sched_barrier(0)
#else
#define SCHED_FENCE()
#endif

// ---------------------------------------------------------------------------
// WMMA helpers (gfx1250, wave32). Fragment layouts per CDNA5 ISA 7.12.2:
//  A (16x32 f16): lane L -> row M = L&15; K = kb8+{0..7} (v0-3) and kb8+16+{0..7}
//                 (v4-7) where kb8 = (L>=16)*8. Two contiguous 16B runs.
//  B (32x16 f16): lane L -> col N = L&15; K order same as A (kb8 = (L>=16)*8).
//                 Two contiguous 16B runs of B^T row-major [N][K].
//  C/D (16x16 f32): lane L -> col N = L&15; VGPR r -> row M = r + (L>=16)*8.
// ---------------------------------------------------------------------------
__device__ __forceinline__ v8f wmma16(v16h a, v16h b, v8f c) {
  return __builtin_amdgcn_wmma_f32_16x16x32_f16(false, a, false, b,
                                                (short)0, c, false, false);
}

__device__ __forceinline__ v16h load_a_frag(const _Float16* tile, int ld) {
  const int lane = threadIdx.x & 31;
  const int m   = lane & 15;
  const int kb8 = (lane >> 4) << 3;
  const h8 lo = *(const h8*)(tile + m * ld + kb8);
  const h8 hi = *(const h8*)(tile + m * ld + kb8 + 16);
  v16h out;
#pragma unroll
  for (int i = 0; i < 8; ++i) { out[i] = lo[i]; out[i + 8] = hi[i]; }
  return out;
}

__device__ __forceinline__ v16h load_b_frag(const _Float16* tileT, int ld) {
  // tileT = B^T, row-major [N][K]
  const int lane = threadIdx.x & 31;
  const int n    = lane & 15;
  const int kb8  = (lane >> 4) << 3;
  const h8 lo = *(const h8*)(tileT + n * ld + kb8);
  const h8 hi = *(const h8*)(tileT + n * ld + kb8 + 16);
  v16h out;
#pragma unroll
  for (int i = 0; i < 8; ++i) { out[i] = lo[i]; out[i + 8] = hi[i]; }
  return out;
}

// ---------------------------------------------------------------------------
// f32 -> f16 conversion, 8 elements per thread
// ---------------------------------------------------------------------------
__global__ void cvt8_kernel(const float* __restrict__ src,
                            _Float16* __restrict__ dst, int n8) {
  int i = blockIdx.x * blockDim.x + threadIdx.x;
  if (i >= n8) return;
  const float4* s = (const float4*)src;
  float4 a = s[2 * i + 0];
  float4 c = s[2 * i + 1];
  h8 o;
  o[0] = (_Float16)a.x; o[1] = (_Float16)a.y;
  o[2] = (_Float16)a.z; o[3] = (_Float16)a.w;
  o[4] = (_Float16)c.x; o[5] = (_Float16)c.y;
  o[6] = (_Float16)c.z; o[7] = (_Float16)c.w;
  ((h8*)dst)[i] = o;
}

// ---------------------------------------------------------------------------
// QKV projection: qkv[8192,1536] = xh[8192,512] @ wh^T + bias
// Wave computes a 16(M) x 64(N) tile. Fully-unrolled K loop with one-stage
// lookahead; sched_barrier keeps prefetch loads above the WMMAs and
// amdgpu_waves_per_eu(4) gives the scheduler register room (verified: waits
// run at loadcnt 14-20 outstanding in the steady state).
// Grid (24, 64) x 256 threads covers 1536 x 8192 exactly -> EXEC all ones.
// ---------------------------------------------------------------------------
__global__ __launch_bounds__(256)
__attribute__((amdgpu_waves_per_eu(4)))
void qkv_kernel(const _Float16* __restrict__ xh, const _Float16* __restrict__ wh,
                const float* __restrict__ bias,
                _Float16* __restrict__ qb, _Float16* __restrict__ kb,
                _Float16* __restrict__ vt) {
  const int lane = threadIdx.x & 31;
  const int w    = threadIdx.x >> 5;
  const int n0   = blockIdx.x * 64;
  const int m0   = (blockIdx.y * 8 + w) * 16;

  v8f zero = 0.f;
  v8f acc[4];
#pragma unroll
  for (int s = 0; s < 4; ++s) acc[s] = zero;

  // prologue: fragments for k0 = 0
  v16h a_c = load_a_frag(xh + m0 * DIMX, DIMX);
  v16h b_c[4];
#pragma unroll
  for (int s = 0; s < 4; ++s)
    b_c[s] = load_b_frag(wh + (n0 + s * 16) * DIMX, DIMX);

#pragma unroll
  for (int k0 = 0; k0 < DIMX - 32; k0 += 32) {
    // prefetch next iteration's fragments (issued before current WMMAs)
    v16h a_n = load_a_frag(xh + m0 * DIMX + k0 + 32, DIMX);
    v16h b_n[4];
#pragma unroll
    for (int s = 0; s < 4; ++s)
      b_n[s] = load_b_frag(wh + (n0 + s * 16) * DIMX + k0 + 32, DIMX);

    SCHED_FENCE();   // loads above stay above the WMMAs below

#pragma unroll
    for (int s = 0; s < 4; ++s) acc[s] = wmma16(a_c, b_c[s], acc[s]);

    a_c = a_n;
#pragma unroll
    for (int s = 0; s < 4; ++s) b_c[s] = b_n[s];
  }
  // epilogue iteration
#pragma unroll
  for (int s = 0; s < 4; ++s) acc[s] = wmma16(a_c, b_c[s], acc[s]);

  const int hi8   = (lane >> 4) << 3;
  const int gr    = m0 + hi8;        // first of 8 consecutive rows in this lane
  const int batch = gr >> 12;        // gr / 4096
  const int srow  = gr & 4095;

#pragma unroll
  for (int s = 0; s < 4; ++s) {
    const int o      = n0 + s * 16 + (lane & 15);
    const int region = o >> 9;       // 0:q 1:k 2:v
    const int c      = o & 511;
    const int hh     = c & 3;
    const int d      = c >> 2;
    const float bv   = bias[o];
    if (region == 2) {
      // v transposed: [b,h,d,n]; 8 consecutive n -> one 16B store
      h8 pk;
#pragma unroll
      for (int r = 0; r < 8; ++r) pk[r] = (_Float16)(acc[s][r] + bv);
      *(h8*)(vt + ((batch * HEADS + hh) * DH + d) * SEQ + srow) = pk;
    } else {
      _Float16* dst = (region == 0 ? qb : kb) +
                      ((batch * HEADS + hh) * SEQ + srow) * DH + d;
      const float sc = (region == 0) ? SCALE_Q : 1.0f;
#pragma unroll
      for (int r = 0; r < 8; ++r)
        dst[r * DH] = (_Float16)((acc[s][r] + bv) * sc);
    }
  }
}

// ---------------------------------------------------------------------------
// Flash attention with cooperative LDS staging.
// All 8 waves of a block share one (b,h) and stream IDENTICAL K/V chunks, so
// each chunk is staged once in LDS (8x less L2 traffic; K/V re-read was the
// roofline limiter). Double-buffered: each thread's one 32B K-slice and one
// 32B V-slice for chunk i+1 are global-loaded at the top of iteration i
// (pinned by sched_barrier), ds_stored after the compute, one
// __syncthreads() per chunk.
// Per wave the math is unchanged: S^T = K·Q^T puts exp'd probabilities in
// the exact A-operand layout for P·V (no transpose).
// 2048 wave-tiles -> 256 blocks x 8 waves, exact cover -> EXEC all ones.
// ---------------------------------------------------------------------------
__global__ __launch_bounds__(256)
void attn_kernel(const _Float16* __restrict__ qb, const _Float16* __restrict__ kb,
                 const _Float16* __restrict__ vt, float* __restrict__ out) {
  __shared__ _Float16 Ks[2][CH * KSTRIDE];   // 2*32*136*2  = 17408 B
  __shared__ _Float16 Vs[2][DH * VSTRIDE];   // 2*128*40*2  = 20480 B

  const int tid   = threadIdx.x;
  const int lane  = tid & 31;
  const int w     = tid >> 5;
  const int tile  = blockIdx.x * 8 + w;
  const int bh    = tile >> 8;          // constant within block (8 | 256)
  const int qt    = tile & 255;         // query tile within sequence
  const int batch = bh >> 2;
  const int h     = bh & 3;

  const _Float16* q = qb + bh * SEQ * DH;
  const _Float16* k = kb + bh * SEQ * DH;
  const _Float16* v = vt + bh * DH * SEQ;

  // cooperative fill mapping: one 32B slice of K and of V per thread
  const int krow = tid >> 3, kcol = (tid & 7) * 16;   // K chunk: 32 x 128
  const int vrow = tid >> 1, vcol = (tid & 1) * 16;   // V chunk: 128 x 32

  // Q fragments (B-operand: B^T row-major [query][dh]) held for whole K loop
  v16h qf[4];
#pragma unroll
  for (int j = 0; j < 4; ++j)
    qf[j] = load_b_frag(q + qt * 16 * DH + j * 32, DH);

  // prologue: stage chunk 0
  {
    v16h kf = *(const v16h*)(k + krow * DH + kcol);
    v16h vf = *(const v16h*)(v + vrow * SEQ + vcol);
    *(v16h*)(&Ks[0][krow * KSTRIDE + kcol]) = kf;
    *(v16h*)(&Vs[0][vrow * VSTRIDE + vcol]) = vf;
  }
  __syncthreads();

  v8f zero = 0.f;
  v8f O[8];
#pragma unroll
  for (int t = 0; t < 8; ++t) O[t] = zero;

  float m = -3.0e38f;
  float l = 0.f;
  const int hi8 = (lane >> 4) << 3;

  int cur = 0;
  for (int kc = 0; kc < SEQ; kc += CH) {
    const int nxt = cur ^ 1;
    // prefetch next chunk from global (clamped on the last iteration: the
    // redundant re-load is uniform, branch-free, and never consumed)
    const int kn = (kc + CH < SEQ) ? (kc + CH) : kc;
    v16h kf = *(const v16h*)(k + (kn + krow) * DH + kcol);
    v16h vf = *(const v16h*)(v + vrow * SEQ + kn + vcol);

    SCHED_FENCE();   // keep global prefetch issue above the compute

    const _Float16* ksh = Ks[cur];
    const _Float16* vsh = Vs[cur];

    // --- S^T = K · Q^T (two 16-key tiles) from LDS ---
    v8f s0 = zero, s1 = zero;
#pragma unroll
    for (int j = 0; j < 4; ++j) {
      v16h a0 = load_a_frag(ksh + j * 32, KSTRIDE);                 // keys 0-15
      v16h a1 = load_a_frag(ksh + 16 * KSTRIDE + j * 32, KSTRIDE);  // keys 16-31
      s0 = wmma16(a0, qf[j], s0);
      s1 = wmma16(a1, qf[j], s1);
    }

    // This lane holds 16 of the 32 key-scores for query (lane&15);
    // lane^16 holds the other 16 -> one xor-16 shuffle completes the row.
    float cm = s0[0];
#pragma unroll
    for (int i = 1; i < 8; ++i) cm = fmaxf(cm, s0[i]);
#pragma unroll
    for (int i = 0; i < 8; ++i) cm = fmaxf(cm, s1[i]);
    cm = fmaxf(cm, __shfl_xor(cm, 16, 32));
    const float mn    = fmaxf(m, cm);
    const float alpha = __expf(m - mn);

    // exp in-register; result is already the A-fragment for P·V
    v16h af;
    float rs = 0.f;
#pragma unroll
    for (int i = 0; i < 8; ++i) {
      float p0 = __expf(s0[i] - mn);
      float p1 = __expf(s1[i] - mn);
      rs += p0 + p1;
      af[i]     = (_Float16)p0;
      af[i + 8] = (_Float16)p1;
    }
    rs += __shfl_xor(rs, 16, 32);
    l = l * alpha + rs;
    m = mn;

    // Broadcast per-query alpha to the O accumulator row layout
    float a8[8];
#pragma unroll
    for (int r = 0; r < 8; ++r) a8[r] = __shfl(alpha, hi8 + r, 32);
#pragma unroll
    for (int t = 0; t < 8; ++t) {
#pragma unroll
      for (int r = 0; r < 8; ++r) O[t][r] *= a8[r];
    }

    // --- O += P · V from LDS ---
#pragma unroll
    for (int t = 0; t < 8; ++t) {
      v16h bv = load_b_frag(vsh + (t * 16) * VSTRIDE, VSTRIDE);
      O[t] = wmma16(af, bv, O[t]);
    }

    // publish next chunk (global loads completed long ago) and swap
    *(v16h*)(&Ks[nxt][krow * KSTRIDE + kcol]) = kf;
    *(v16h*)(&Vs[nxt][vrow * VSTRIDE + vcol]) = vf;
    __syncthreads();
    cur = nxt;
  }

  // Normalize rows and store: out[b, row, h*128 + d]
  const float linv = 1.f / l;
  float li[8];
#pragma unroll
  for (int r = 0; r < 8; ++r) li[r] = __shfl(linv, hi8 + r, 32);

  const int n15 = lane & 15;
#pragma unroll
  for (int t = 0; t < 8; ++t) {
#pragma unroll
    for (int r = 0; r < 8; ++r) {
      const int row = qt * 16 + hi8 + r;
      out[(batch * SEQ + row) * DIMX + h * DH + t * 16 + n15] = O[t][r] * li[r];
    }
  }
}

// ---------------------------------------------------------------------------
// Launch. Workspace layout (bytes):
//   [0,8M)    xh   8192*512 f16
//   [8M,10M)  wh   1536*512 f16
//   [10M,18M) q    [b,h,n,dh] f16
//   [18M,26M) k    [b,h,n,dh] f16
//   [26M,34M) vt   [b,h,dh,n] f16
// ---------------------------------------------------------------------------
extern "C" void kernel_launch(void* const* d_in, const int* in_sizes, int n_in,
                              void* d_out, int out_size, void* d_ws, size_t ws_size,
                              hipStream_t stream) {
  const float* x = (const float*)d_in[0];
  const float* W = (const float*)d_in[1];
  const float* b = (const float*)d_in[2];
  float* out = (float*)d_out;

  char* ws = (char*)d_ws;
  const size_t MB = 1024 * 1024;
  _Float16* xh = (_Float16*)(ws + 0 * MB);
  _Float16* wh = (_Float16*)(ws + 8 * MB);
  _Float16* qb = (_Float16*)(ws + 10 * MB);
  _Float16* kb = (_Float16*)(ws + 18 * MB);
  _Float16* vt = (_Float16*)(ws + 26 * MB);

  const int nx8 = (ROWS * DIMX) / 8;   // 524288
  const int nw8 = (NOUT * DIMX) / 8;   // 98304
  cvt8_kernel<<<(nx8 + 255) / 256, 256, 0, stream>>>(x, xh, nx8);
  cvt8_kernel<<<(nw8 + 255) / 256, 256, 0, stream>>>(W, wh, nw8);

  qkv_kernel<<<dim3(NOUT / 64, ROWS / (16 * 8)), 256, 0, stream>>>(
      xh, wh, b, qb, kb, vt);

  attn_kernel<<<(BATCH * HEADS * (SEQ / 16)) / 8, 256, 0, stream>>>(
      qb, kb, vt, out);
}
